// MultiViewSRRNet_V9_82755429859616
// MI455X (gfx1250) — matrix operand order repaired and checked by reading, not compiled, not run on hardware
//
#include <hip/hip_runtime.h>

typedef __attribute__((ext_vector_type(16))) _Float16 v16h;
typedef __attribute__((ext_vector_type(8)))  _Float16 v8h;
typedef __attribute__((ext_vector_type(8)))  float    v8f;

#define B_    16
#define VIEWS 3
#define P_    48
#define HW_   1024
#define DPc   544      // 515 padded to multiple of 32
#define KC    128      // clusters / nodes per view
#define Fd    128      // feature dim
#define NLAT  32

// ---------------------------------------------------------------- wave utils
__device__ inline float wave_sum(float s){
#pragma unroll
  for (int o=1;o<32;o<<=1) s += __shfl_xor(s,o,32);
  return s;
}
__device__ inline float wave_max(float s){
#pragma unroll
  for (int o=1;o<32;o<<=1) s = fmaxf(s,__shfl_xor(s,o,32));
  return s;
}

// ------------------------------------------------- WMMA fragment loaders (f16)
// A (16x32, row-major, lda halves): lane l -> row m=l&15, hi=l>>4
//   halves[0..7]  = A[m][k0 + hi*8 + 0..7]
//   halves[8..15] = A[m][k0 + 16 + hi*8 + 0..7]
// Full-tile loaders: no bounds checks, no exec manipulation (wave-uniform use).
__device__ inline v16h frag_a_full(const _Float16* A, int lda, int row0, int k0, int lane){
  int m  = row0 + (lane & 15);
  int hi = lane >> 4;
  const _Float16* pr = A + (long long)m*lda + k0 + hi*8;
  __builtin_prefetch(pr + 64, 0, 1);            // 2 K-steps ahead -> global_prefetch_b8
  v8h lo = *(const v8h*)pr;
  v8h hh = *(const v8h*)(pr + 16);
  v16h f;
#pragma unroll
  for (int i=0;i<8;i++){ f[i]=lo[i]; f[i+8]=hh[i]; }
  return f;
}
// B (32x16) loaded from Bt (N x Kd row-major = W^T): lane l -> col n=l&15,
//   halves[0..7] = Bt[n][k0 + (l>>4)*8 + 0..7], halves[8..15] = Bt[n][k0 + 16 + (l>>4)*8 + 0..7]
__device__ inline v16h frag_b_full(const _Float16* Bt, int ldb, int col0, int k0, int lane){
  int n  = col0 + (lane & 15);
  int kb = (lane >> 4) * 8;
  const _Float16* pr = Bt + (long long)n*ldb + k0 + kb;
  v8h lo = *(const v8h*)pr;
  v8h hh = *(const v8h*)(pr + 16);
  v16h f;
#pragma unroll
  for (int i=0;i<8;i++){ f[i]=lo[i]; f[i+8]=hh[i]; }
  return f;
}
// Guarded variants (only taken on genuinely partial tiles, e.g. M=4 fusion GEMMs)
__device__ inline v16h frag_a_guard(const _Float16* A, int lda, int M, int row0, int k0, int lane){
  int m  = row0 + (lane & 15);
  int hi = lane >> 4;
  v16h f;
  if (m < M){
    const _Float16* pr = A + (long long)m*lda + k0 + hi*8;
    v8h lo = *(const v8h*)pr;
    v8h hh = *(const v8h*)(pr + 16);
#pragma unroll
    for (int i=0;i<8;i++){ f[i]=lo[i]; f[i+8]=hh[i]; }
  } else {
#pragma unroll
    for (int i=0;i<16;i++) f[i] = (_Float16)0.f;
  }
  return f;
}
__device__ inline v16h frag_b_guard(const _Float16* Bt, int ldb, int N, int col0, int k0, int lane){
  int n  = col0 + (lane & 15);
  int kb = (lane >> 4) * 8;
  v16h f;
  if (n < N){
    const _Float16* pr = Bt + (long long)n*ldb + k0 + kb;
    v8h lo = *(const v8h*)pr;
    v8h hh = *(const v8h*)(pr + 16);
#pragma unroll
    for (int i=0;i<8;i++){ f[i]=lo[i]; f[i+8]=hh[i]; }
  } else {
#pragma unroll
    for (int i=0;i<16;i++) f[i] = (_Float16)0.f;
  }
  return f;
}

__device__ inline v8f wmma16(v16h a, v16h b, v8f c){
  return __builtin_amdgcn_wmma_f32_16x16x32_f16(false, a, false, b, (short)0, c, false, false);
}

// ---------------------------------------------------------------- generic GEMM
// C[M x N](f32) = act( A[M x Kd] @ Bt[N x Kd]^T * scale + bias + resid ), f16 mirror optional.
// z -> (b = z/nh, h = z%nh) two-level batching for per-head submatrices.
struct GemmP {
  const _Float16* A; const _Float16* Bt; float* C; _Float16* Ch;
  const float* bias; const float* resid;
  int lda, ldb, ldc, ldch, ldr;
  int M, N, Kd, nh, act;              // act: 0 none, 1 relu, 2 tanh
  float scale;
  long long sAb, sAh, sBb, sBh, sCb, sCh_, sChb, sChh, sBiash, sRb;
};

__global__ __launch_bounds__(32) void gemm_kernel(GemmP p){
  int z = blockIdx.z; int b = z / p.nh; int h = z - b*p.nh;
  const _Float16* A  = p.A  + b*p.sAb + h*p.sAh;
  const _Float16* Bt = p.Bt + b*p.sBb + h*p.sBh;
  int row0 = blockIdx.y << 4, col0 = blockIdx.x << 4;
  int lane = threadIdx.x;
  v8f acc = {};
  // Wave-uniform full-tile predicate: fast path has no per-lane guards at all.
  if (row0 + 16 <= p.M && col0 + 16 <= p.N){
    for (int k0 = 0; k0 < p.Kd; k0 += 32){
      v16h af = frag_a_full(A,  p.lda, row0, k0, lane);
      v16h bf = frag_b_full(Bt, p.ldb, col0, k0, lane);
      acc = wmma16(af, bf, acc);
    }
  } else {
    for (int k0 = 0; k0 < p.Kd; k0 += 32){
      v16h af = frag_a_guard(A,  p.lda, p.M, row0, k0, lane);
      v16h bf = frag_b_guard(Bt, p.ldb, p.N, col0, k0, lane);
      acc = wmma16(af, bf, acc);
    }
  }
  int n  = col0 + (lane & 15);
  int hi = lane >> 4;
  if (n >= p.N) return;
  float bv = p.bias ? p.bias[p.sBiash*h + n] : 0.f;
#pragma unroll
  for (int r = 0; r < 8; ++r){
    int m = row0 + hi*8 + r;
    if (m >= p.M) break;
    float v = acc[r]*p.scale + bv;
    if (p.resid) v += p.resid[b*p.sRb + (long long)m*p.ldr + n];
    if (p.act == 1) v = fmaxf(v, 0.f);
    else if (p.act == 2) v = tanhf(v);
    if (p.C)  p.C [b*p.sCb  + h*p.sCh_ + (long long)m*p.ldc  + n] = v;
    if (p.Ch) p.Ch[b*p.sChb + h*p.sChh + (long long)m*p.ldch + n] = (_Float16)v;
  }
}

static GemmP gp(){ GemmP p{}; p.scale = 1.f; p.nh = 1; return p; }
static void LG(hipStream_t st, const GemmP& p, int zb){
  dim3 grid((p.N+15)/16, (p.M+15)/16, zb);
  gemm_kernel<<<grid, 32, 0, st>>>(p);
}

// ------------------------------------------------------------- kmeans kernels
__global__ void build_comb(const float* fs, const float* fc, const float* fa, _Float16* X, long long total){
  long long idx = (long long)blockIdx.x*256 + threadIdx.x;
  if (idx >= total) return;
  int d = (int)(idx % DPc);
  long long t = idx / DPc;
  int n = (int)(t & 1023);
  int p = (int)(t >> 10);
  int b = p / 3, v = p % 3;
  float val;
  if (d < 512){
    const float* f = (v==0) ? fs : ((v==1) ? fc : fa);
    val = f[((long long)b*512 + d)*1024 + n];
  } else if (d == 512) val = 2.5f;                                 // z=0.5*5
  else if (d == 513)   val = ((n >> 5) * (1.f/31.f)) * 5.f;        // gy*5
  else if (d == 514)   val = ((n & 31) * (1.f/31.f)) * 5.f;        // gx*5
  else val = 0.f;
  X[idx] = (_Float16)val;
}

__global__ void cent_init(const _Float16* X, _Float16* cent){
  int pk = blockIdx.x; int p = pk >> 7; int k = pk & 127;
  const _Float16* src = X + ((long long)p*HW_ + k*8)*DPc;
  _Float16* dst = cent + (long long)pk*DPc;
  for (int d = threadIdx.x; d < DPc; d += 256) dst[d] = src[d];
}

__global__ __launch_bounds__(32) void cnorm_kernel(const _Float16* cent, float* cn){
  int pk = blockIdx.x; int lane = threadIdx.x;
  const _Float16* cr = cent + (long long)pk*DPc;
  float s = 0.f;
  for (int d = lane; d < DPc; d += 32){ float v = (float)cr[d]; s += v*v; }
  s = wave_sum(s);
  if (lane == 0) cn[pk] = s;
}

// fused WMMA distance GEMM + cross-lane argmin: label = argmin_c (|c|^2 - 2 x.c)
// All shapes are exact multiples of 16 -> unguarded fragment loads only.
__global__ __launch_bounds__(32) void kmeans_assign(const _Float16* X, const _Float16* C,
                                                    const float* cnorm, int* labels){
  int p = blockIdx.y; int row0 = blockIdx.x << 4;
  int lane = threadIdx.x;
  const _Float16* A  = X + (long long)p*HW_*DPc;
  const _Float16* Bt = C + (long long)p*KC*DPc;
  const float* cn = cnorm + p*KC;
  float bestv[8]; int besti[8];
#pragma unroll
  for (int r=0;r<8;r++){ bestv[r] = 3.4e38f; besti[r] = 0; }
  for (int c0 = 0; c0 < KC; c0 += 16){
    v8f acc = {};
    for (int k0 = 0; k0 < DPc; k0 += 32)
      acc = wmma16(frag_a_full(A, DPc, row0, k0, lane), frag_b_full(Bt, DPc, c0, k0, lane), acc);
    int n = c0 + (lane & 15);
    float cnv = cn[n];
#pragma unroll
    for (int r=0;r<8;r++){
      float d = cnv - 2.f*acc[r];
      int ci = n;
#pragma unroll
      for (int off=1; off<16; off<<=1){
        float ov = __shfl_xor(d, off, 32);
        int   oi = __shfl_xor(ci, off, 32);
        if (ov < d || (ov == d && oi < ci)){ d = ov; ci = oi; }
      }
      if (d < bestv[r] || (d == bestv[r] && ci < besti[r])){ bestv[r] = d; besti[r] = ci; }
    }
  }
  if ((lane & 15) == 0){
    int hi = lane >> 4;
#pragma unroll
    for (int r=0;r<8;r++) labels[p*HW_ + row0 + hi*8 + r] = besti[r];
  }
}

// deterministic gather-reduction centroid update (no atomics)
__global__ void cent_update(const _Float16* X, const int* labels, _Float16* cent){
  int k = blockIdx.x, p = blockIdx.y, tid = threadIdx.x;
  float a0=0.f, a1=0.f, a2=0.f; int cnt=0;
  const int* lab = labels + p*HW_;
  for (int n = 0; n < HW_; ++n){
    if (lab[n] == k){
      const _Float16* xr = X + ((long long)p*HW_ + n)*DPc;
      a0 += (float)xr[tid];
      a1 += (float)xr[tid + 256];
      if (tid + 512 < DPc) a2 += (float)xr[tid + 512];
      cnt++;
    }
  }
  float inv = 1.f / fmaxf((float)cnt, 1.f);
  _Float16* cr = cent + ((long long)p*KC + k)*DPc;
  cr[tid]       = (_Float16)(a0*inv);
  cr[tid + 256] = (_Float16)(a1*inv);
  if (tid + 512 < DPc) cr[tid + 512] = (_Float16)(a2*inv);
}

// ------------------------------------------------------------- graph building
__global__ void build_adj_raw(const _Float16* cent, float* raw){
  int p = blockIdx.x; int i = threadIdx.x;  // 128
  __shared__ float cc[KC][3];
  for (int t=0;t<3;t++) cc[i][t] = (float)cent[((long long)p*KC + i)*DPc + 512 + t];
  __syncthreads();
  float ci0 = cc[i][0], ci1 = cc[i][1], ci2 = cc[i][2];
  unsigned long long s0 = 0ull, s1 = 0ull;
  for (int t=0;t<8;t++){
    float best = 3.4e38f; int bj = 0;
    for (int j=0;j<KC;j++){
      bool used = (j < 64) ? ((s0 >> j) & 1ull) : ((s1 >> (j-64)) & 1ull);
      if (used) continue;
      float d0 = ci0-cc[j][0], d1 = ci1-cc[j][1], d2 = ci2-cc[j][2];
      float d = d0*d0 + d1*d1 + d2*d2;
      if (d < best){ best = d; bj = j; }
    }
    if (bj < 64) s0 |= 1ull << bj; else s1 |= 1ull << (bj-64);
  }
  float* row = raw + ((long long)p*KC + i)*KC;
  for (int j=0;j<KC;j++){
    bool sel = (j < 64) ? ((s0 >> j) & 1ull) : ((s1 >> (j-64)) & 1ull);
    row[j] = sel ? 1.f : 0.f;
  }
}

__global__ void adj_sym(const float* raw, float* adj, long long total){
  long long idx = (long long)blockIdx.x*256 + threadIdx.x;
  if (idx >= total) return;
  int j = (int)(idx % KC);
  long long t = idx / KC;
  int i = (int)(t % KC);
  int p = (int)(t / KC);
  float a = raw[((long long)p*KC + i)*KC + j] + raw[((long long)p*KC + j)*KC + i];
  adj[idx] = (a > 0.f) ? 1.f : 0.f;
}

// --------------------------------------------------------------- small kernels
__global__ void copy3bias(const float* b0, const float* b1, const float* b2, float* out){
  int t = threadIdx.x;
  if (t < 128) out[t] = b0[t];
  else if (t < 256) out[t] = b1[t-128];
  else out[t] = b2[t-256];
}

__global__ void cast_t_kernel(const float* W, int Kd, int Nfull, int col0, int ncols,
                              _Float16* out, int KdP, long long total){
  long long idx = (long long)blockIdx.x*256 + threadIdx.x;
  if (idx >= total) return;
  int k = (int)(idx % KdP);
  int n = (int)(idx / KdP);
  out[idx] = (k < Kd) ? (_Float16)W[(long long)k*Nfull + col0 + n] : (_Float16)0.f;
}

__global__ void headT_kernel(const _Float16* in, _Float16* out, int Mkv, int H, int C, long long total){
  long long idx = (long long)blockIdx.x*256 + threadIdx.x;
  if (idx >= total) return;
  int j = (int)(idx % Mkv); long long t = idx / Mkv;
  int c = (int)(t % C); t /= C;
  int h = (int)(t % H);
  int b = (int)(t / H);
  out[idx] = in[((long long)b*Mkv + j)*(H*C) + h*C + c];
}

__global__ void bcast_lat(const float* tok, float* l32, _Float16* l16, long long total){
  long long i = (long long)blockIdx.x*256 + threadIdx.x;
  if (i >= total) return;
  float v = tok[i % (NLAT*Fd)];
  l32[i] = v; l16[i] = (_Float16)v;
}

__global__ __launch_bounds__(32) void softmax_rows(const float* x, _Float16* o, int L){
  long long row = blockIdx.x; int lane = threadIdx.x;
  const float* xr = x + row*L;
  float mx = -3.4e38f;
  for (int j=lane; j<L; j+=32) mx = fmaxf(mx, xr[j]);
  mx = wave_max(mx);
  float s = 0.f;
  for (int j=lane; j<L; j+=32) s += __expf(xr[j]-mx);
  s = wave_sum(s);
  float inv = 1.f/s;
  for (int j=lane; j<L; j+=32) o[row*L + j] = (_Float16)(__expf(xr[j]-mx)*inv);
}

__global__ __launch_bounds__(32) void ln128(const float* x, const float* resid, const float* g,
                                            const float* be, float* o32, _Float16* o16){
  long long row = blockIdx.x; int lane = threadIdx.x;
  const float* xr = x + row*Fd; const float* rr = resid + row*Fd;
  float v[4]; float s = 0.f;
#pragma unroll
  for (int t=0;t<4;t++){ int c = lane + 32*t; v[t] = xr[c] + rr[c]; s += v[t]; }
  s = wave_sum(s);
  float mean = s*(1.f/128.f);
  float q = 0.f;
#pragma unroll
  for (int t=0;t<4;t++){ float d = v[t]-mean; q += d*d; }
  q = wave_sum(q);
  float inv = rsqrtf(q*(1.f/128.f) + 1e-5f);
#pragma unroll
  for (int t=0;t<4;t++){
    int c = lane + 32*t;
    float y = (v[t]-mean)*inv*g[c] + be[c];
    if (o32) o32[row*Fd + c] = y;
    if (o16) o16[row*Fd + c] = (_Float16)y;
  }
}

// ----------------------------------------------------------------- GAT kernels
__global__ __launch_bounds__(32) void gat_asd(const float* hb, const float* asrc, const float* adst,
                                              float* a_s, float* a_d, int N, int HC, int C){
  int n = blockIdx.x, h = blockIdx.y, b = blockIdx.z; int H = gridDim.y;
  int lane = threadIdx.x;
  const float* hr = hb + ((long long)b*N + n)*HC + h*C;
  float s = 0.f, d = 0.f;
  for (int c = lane; c < C; c += 32){ float hv = hr[c]; s += hv*asrc[h*C+c]; d += hv*adst[h*C+c]; }
  s = wave_sum(s); d = wave_sum(d);
  if (lane == 0){ a_s[((long long)b*N+n)*H+h] = s; a_d[((long long)b*N+n)*H+h] = d; }
}

__global__ __launch_bounds__(32) void gat_alpha(const float* a_s, const float* a_d, const float* adj,
                                                long long adj_bstride, _Float16* alpha, int N, int H){
  int i = blockIdx.x, h = blockIdx.y, b = blockIdx.z;
  int lane = threadIdx.x;
  const float* arow = adj + (long long)b*adj_bstride + (long long)i*N;
  float adi = a_d[((long long)b*N + i)*H + h];
  float mx = -3.4e38f;
  for (int j=lane; j<N; j+=32){
    if (arow[j] > 0.f || j == i){
      float e = a_s[((long long)b*N + j)*H + h] + adi;
      e = (e > 0.f) ? e : 0.2f*e;
      mx = fmaxf(mx, e);
    }
  }
  mx = wave_max(mx);
  float sum = 0.f;
  for (int j=lane; j<N; j+=32){
    if (arow[j] > 0.f || j == i){
      float e = a_s[((long long)b*N + j)*H + h] + adi;
      e = (e > 0.f) ? e : 0.2f*e;
      sum += __expf(e - mx);
    }
  }
  sum = wave_sum(sum);
  float inv = 1.f/sum;
  _Float16* orow = alpha + ((long long)(b*H + h)*N + i)*N;
  for (int j=lane; j<N; j+=32){
    float o = 0.f;
    if (arow[j] > 0.f || j == i){
      float e = a_s[((long long)b*N + j)*H + h] + adi;
      e = (e > 0.f) ? e : 0.2f*e;
      o = __expf(e - mx)*inv;
    }
    orow[j] = (_Float16)o;
  }
}

// ---------------------------------------------------------------- pooling etc.
__global__ __launch_bounds__(32) void rowdot(const float* x, long long xbstride, const float* w,
                                             const float* bsc, float* out, int N, int Csz){
  int bn = blockIdx.x; int b = bn / N; int n = bn % N;
  int lane = threadIdx.x;
  const float* xr = x + (long long)b*xbstride + (long long)n*Csz;
  float s = 0.f;
  for (int c = lane; c < Csz; c += 32) s += xr[c]*w[c];
  s = wave_sum(s);
  if (lane == 0) out[bn] = s + bsc[0];
}

__global__ void topk_desc(const float* score, int N, int k, int* idx){
  if (threadIdx.x != 0) return;
  int b = blockIdx.x;
  bool used[KC];
  for (int j=0;j<N;j++) used[j] = false;
  for (int t=0;t<k;t++){
    float best = -3.4e38f; int bi = 0;
    for (int j=0;j<N;j++)
      if (!used[j] && score[b*N+j] > best){ best = score[b*N+j]; bi = j; }
    used[bi] = true;
    idx[b*k + t] = bi;
  }
}

__global__ void gather_rows(const float* x, long long xbstride, const int* idx, int k, int Csz,
                            float* out, _Float16* out16){
  int bi = blockIdx.x; int b = bi / k; int i = bi % k;
  int src = idx[b*k + i];
  const float* xr = x + (long long)b*xbstride + (long long)src*Csz;
  for (int c = threadIdx.x; c < Csz; c += 256){
    float v = xr[c];
    out[((long long)b*k + i)*Csz + c] = v;
    out16[((long long)b*k + i)*Csz + c] = (_Float16)v;
  }
}

__global__ void gather_adj(const float* adj, long long adj_bstride, const int* idx, int k, int N, float* out){
  int bi = blockIdx.x; int b = bi / k; int i = bi % k;
  int j = threadIdx.x;
  if (j >= k) return;
  int si = idx[b*k + i], sj = idx[b*k + j];
  out[((long long)b*k + i)*k + j] = adj[(long long)b*adj_bstride + (long long)si*N + sj];
}

__global__ void gate_finish(const float* t, const float* W2, const float* b2,
                            const float* x3, float* h){
  int b = blockIdx.x; int tid = threadIdx.x;  // 128
  __shared__ float s[32];
  __shared__ float wv[32];
  if (tid < 32){
    float acc = b2[0];
    const float* tr = t + ((long long)b*32 + tid)*64;
    for (int c=0;c<64;c++) acc += tr[c]*W2[c];
    s[tid] = acc;
  }
  __syncthreads();
  if (tid == 0){
    float mx = s[0];
    for (int n=1;n<32;n++) mx = fmaxf(mx, s[n]);
    float sum = 0.f;
    for (int n=0;n<32;n++){ wv[n] = __expf(s[n]-mx); sum += wv[n]; }
    float inv = 1.f/sum;
    for (int n=0;n<32;n++) wv[n] *= inv;
  }
  __syncthreads();
  float acc = 0.f;
  for (int n=0;n<32;n++) acc += wv[n]*x3[((long long)b*32 + n)*Fd + tid];
  h[b*Fd + tid] = acc;
}

// ------------------------------------------------------------------- fusion
__global__ void build_xfu(const float* cls, const float* h0, const float* h1, const float* h2,
                          float* x32, _Float16* x16){
  int b = blockIdx.x; int c = threadIdx.x;  // 128
  float v0 = cls[c];
  float v1 = h0[b*Fd + c], v2 = h1[b*Fd + c], v3 = h2[b*Fd + c];
  long long o = (long long)b*4*Fd;
  x32[o + 0*Fd + c] = v0; x16[o + 0*Fd + c] = (_Float16)v0;
  x32[o + 1*Fd + c] = v1; x16[o + 1*Fd + c] = (_Float16)v1;
  x32[o + 2*Fd + c] = v2; x16[o + 2*Fd + c] = (_Float16)v2;
  x32[o + 3*Fd + c] = v3; x16[o + 3*Fd + c] = (_Float16)v3;
}

__global__ __launch_bounds__(32) void fusion_attn(const _Float16* q, const _Float16* k,
                                                  const _Float16* v, _Float16* o){
  int b = blockIdx.x; int lane = threadIdx.x;
  int h = lane >> 2; int i = lane & 3;  // 8 heads x 4 rows, dh=16
  const _Float16* qb = q + (long long)b*512;
  const _Float16* kb = k + (long long)b*512;
  const _Float16* vb = v + (long long)b*512;
  float sc[4]; float mx = -3.4e38f;
#pragma unroll
  for (int j=0;j<4;j++){
    float s = 0.f;
    for (int c=0;c<16;c++) s += (float)qb[i*Fd + h*16 + c]*(float)kb[j*Fd + h*16 + c];
    sc[j] = s*0.25f;  // 1/sqrt(16)
    mx = fmaxf(mx, sc[j]);
  }
  float sum = 0.f;
#pragma unroll
  for (int j=0;j<4;j++){ sc[j] = __expf(sc[j]-mx); sum += sc[j]; }
  float inv = 1.f/sum;
  for (int c=0;c<16;c++){
    float a = 0.f;
#pragma unroll
    for (int j=0;j<4;j++) a += sc[j]*inv*(float)vb[j*Fd + h*16 + c];
    o[(long long)b*512 + i*Fd + h*16 + c] = (_Float16)a;
  }
}

__global__ __launch_bounds__(32) void cls_head(const float* z, const float* g, const float* be,
                                               const float* W, const float* bo, float* out){
  int b = blockIdx.x; int lane = threadIdx.x;
  const float* x = z + (long long)b*4*Fd;  // row 0
  float v[4]; float s = 0.f;
#pragma unroll
  for (int t=0;t<4;t++){ v[t] = x[lane + 32*t]; s += v[t]; }
  s = wave_sum(s);
  float mean = s*(1.f/128.f);
  float q = 0.f;
#pragma unroll
  for (int t=0;t<4;t++){ float d = v[t]-mean; q += d*d; }
  q = wave_sum(q);
  float inv = rsqrtf(q*(1.f/128.f) + 1e-5f);
  float xn[4];
#pragma unroll
  for (int t=0;t<4;t++){ int c = lane + 32*t; xn[t] = (v[t]-mean)*inv*g[c] + be[c]; }
  for (int j=0;j<3;j++){
    float part = 0.f;
#pragma unroll
    for (int t=0;t<4;t++){ int c = lane + 32*t; part += xn[t]*W[c*3 + j]; }
    part = wave_sum(part);
    if (lane == 0) out[b*3 + j] = part + bo[j];
  }
}

__global__ __launch_bounds__(32) void aux_head(const float* lat, const float* g, const float* be,
                                               const float* W, const float* bo, float* out){
  int b = blockIdx.x; int lane = threadIdx.x;
  float v[4]; float s = 0.f;
#pragma unroll
  for (int t=0;t<4;t++){
    int c = lane + 32*t;
    float acc = 0.f;
    for (int n=0;n<NLAT;n++) acc += lat[((long long)b*NLAT + n)*Fd + c];
    v[t] = acc*(1.f/32.f); s += v[t];
  }
  s = wave_sum(s);
  float mean = s*(1.f/128.f);
  float q = 0.f;
#pragma unroll
  for (int t=0;t<4;t++){ float d = v[t]-mean; q += d*d; }
  q = wave_sum(q);
  float inv = rsqrtf(q*(1.f/128.f) + 1e-5f);
  float xn[4];
#pragma unroll
  for (int t=0;t<4;t++){ int c = lane + 32*t; xn[t] = (v[t]-mean)*inv*g[c] + be[c]; }
  for (int j=0;j<3;j++){
    float part = 0.f;
#pragma unroll
    for (int t=0;t<4;t++){ int c = lane + 32*t; part += xn[t]*W[c*3 + j]; }
    part = wave_sum(part);
    if (lane == 0) out[b*3 + j] = part + bo[j];
  }
}

// ------------------------------------------------------------- host-side glue
static void castT(hipStream_t st, const float* W, int Kd, int Nfull, int col0, int ncols,
                  _Float16* out, int KdP){
  long long total = (long long)ncols*KdP;
  cast_t_kernel<<<(int)((total+255)/256), 256, 0, st>>>(W, Kd, Nfull, col0, ncols, out, KdP, total);
}

struct MhaScratch {
  _Float16 *wq,*wk,*wv,*wo,*q16,*k16,*v16,*vT,*att16,*o16; float* scores;
};

static void run_mha(hipStream_t st, int Mq, int Mkv,
                    const _Float16* qin, long long qstride,
                    const _Float16* kvin, long long kvstride,
                    const float* Wqkv, const float* bqkv, const float* Wo, const float* bo,
                    MhaScratch& ms, float* attn_out){
  const int H = 4, dh = 32;
  castT(st, Wqkv, 128, 384, 0,   128, ms.wq, 128);
  castT(st, Wqkv, 128, 384, 128, 128, ms.wk, 128);
  castT(st, Wqkv, 128, 384, 256, 128, ms.wv, 128);
  castT(st, Wo,   128, 128, 0,   128, ms.wo, 128);
  { GemmP p = gp(); p.M=Mq; p.N=128; p.Kd=128;
    p.A=qin; p.lda=128; p.sAb=qstride;
    p.Bt=ms.wq; p.ldb=128;
    p.Ch=ms.q16; p.ldch=128; p.sChb=(long long)Mq*128;
    p.bias=bqkv; LG(st,p,B_); }
  { GemmP p = gp(); p.M=Mkv; p.N=128; p.Kd=128;
    p.A=kvin; p.lda=128; p.sAb=kvstride;
    p.Bt=ms.wk; p.ldb=128;
    p.Ch=ms.k16; p.ldch=128; p.sChb=(long long)Mkv*128;
    p.bias=bqkv+128; LG(st,p,B_); }
  { GemmP p = gp(); p.M=Mkv; p.N=128; p.Kd=128;
    p.A=kvin; p.lda=128; p.sAb=kvstride;
    p.Bt=ms.wv; p.ldb=128;
    p.Ch=ms.v16; p.ldch=128; p.sChb=(long long)Mkv*128;
    p.bias=bqkv+256; LG(st,p,B_); }
  { long long tot = (long long)B_*H*dh*Mkv;
    headT_kernel<<<(int)((tot+255)/256), 256, 0, st>>>(ms.v16, ms.vT, Mkv, H, dh, tot); }
  { GemmP p = gp(); p.nh=H; p.M=Mq; p.N=Mkv; p.Kd=dh;
    p.A=ms.q16; p.lda=128; p.sAb=(long long)Mq*128; p.sAh=dh;
    p.Bt=ms.k16; p.ldb=128; p.sBb=(long long)Mkv*128; p.sBh=dh;
    p.C=ms.scores; p.ldc=Mkv; p.sCb=(long long)H*Mq*Mkv; p.sCh_=(long long)Mq*Mkv;
    p.scale=0.1767766953f; LG(st,p,B_*H); }
  softmax_rows<<<B_*H*Mq, 32, 0, st>>>(ms.scores, ms.att16, Mkv);
  { GemmP p = gp(); p.nh=H; p.M=Mq; p.N=dh; p.Kd=Mkv;
    p.A=ms.att16; p.lda=Mkv; p.sAb=(long long)H*Mq*Mkv; p.sAh=(long long)Mq*Mkv;
    p.Bt=ms.vT; p.ldb=Mkv; p.sBb=(long long)H*dh*Mkv; p.sBh=(long long)dh*Mkv;
    p.Ch=ms.o16; p.ldch=128; p.sChb=(long long)Mq*128; p.sChh=dh;
    LG(st,p,B_*H); }
  { GemmP p = gp(); p.M=Mq; p.N=128; p.Kd=128;
    p.A=ms.o16; p.lda=128; p.sAb=(long long)Mq*128;
    p.Bt=ms.wo; p.ldb=128;
    p.C=attn_out; p.ldc=128; p.sCb=(long long)Mq*128;
    p.bias=bo; LG(st,p,B_); }
}

struct GatScratch {
  _Float16* Wt; float* hbuf; _Float16* h16; _Float16* hT;
  float* a_s; float* a_d; _Float16* alpha;
};

static void run_gat(hipStream_t st, int N, int Cin, int Cout, int H,
                    const _Float16* x16, long long xstride,
                    const float* cW, const float* asrc, const float* adst, const float* cb,
                    const float* adj, long long adj_bstride, int act,
                    GatScratch& gs, float* xout, _Float16* xout16){
  int C = Cout / H;
  castT(st, cW, Cin, Cout, 0, Cout, gs.Wt, Cin);
  { GemmP p = gp(); p.M=N; p.N=Cout; p.Kd=Cin;
    p.A=x16; p.lda=Cin; p.sAb=xstride;
    p.Bt=gs.Wt; p.ldb=Cin;
    p.C=gs.hbuf; p.ldc=Cout; p.sCb=(long long)N*Cout;
    p.Ch=gs.h16; p.ldch=Cout; p.sChb=(long long)N*Cout;
    LG(st,p,B_); }
  gat_asd<<<dim3(N,H,B_), 32, 0, st>>>(gs.hbuf, asrc, adst, gs.a_s, gs.a_d, N, Cout, C);
  gat_alpha<<<dim3(N,H,B_), 32, 0, st>>>(gs.a_s, gs.a_d, adj, adj_bstride, gs.alpha, N, H);
  { long long tot = (long long)B_*H*C*N;
    headT_kernel<<<(int)((tot+255)/256), 256, 0, st>>>(gs.h16, gs.hT, N, H, C, tot); }
  { GemmP p = gp(); p.nh=H; p.M=N; p.N=C; p.Kd=N;
    p.A=gs.alpha; p.lda=N; p.sAb=(long long)H*N*N; p.sAh=(long long)N*N;
    p.Bt=gs.hT; p.ldb=N; p.sBb=(long long)H*C*N; p.sBh=(long long)C*N;
    p.C=xout; p.ldc=Cout; p.sCb=(long long)N*Cout; p.sCh_=C;
    p.Ch=xout16; p.ldch=Cout; p.sChb=(long long)N*Cout; p.sChh=C;
    p.bias=cb; p.sBiash=C; p.act=act;
    LG(st,p,B_*H); }
}

static void run_pool(hipStream_t st, int N, int Cin, int k,
                     const float* x, long long xbstride, const float* adj, long long adj_bstride,
                     const float* pW, const float* pb,
                     float* score_out, int* idx, float* nx, _Float16* nx16, float* nadj){
  rowdot<<<B_*N, 32, 0, st>>>(x, xbstride, pW, pb, score_out, N, Cin);
  topk_desc<<<B_, 1, 0, st>>>(score_out, N, k, idx);
  gather_rows<<<B_*k, 256, 0, st>>>(x, xbstride, idx, k, Cin, nx, nx16);
  gather_adj<<<B_*k, 64, 0, st>>>(adj, adj_bstride, idx, k, N, nadj);
}

// ---------------------------------------------------------------- entry point
extern "C" void kernel_launch(void* const* d_in, const int* in_sizes, int n_in,
                              void* d_out, int out_size, void* d_ws, size_t ws_size,
                              hipStream_t stream){
  (void)in_sizes; (void)n_in; (void)out_size; (void)ws_size;
  hipStream_t st = stream;
  const float* fsag = (const float*)d_in[0];
  const float* fcor = (const float*)d_in[1];
  const float* faxi = (const float*)d_in[2];
  auto Pp = [&](int i)->const float* { return (const float*)d_in[3+i]; };
  float* dout = (float*)d_out;

  char* wsb = (char*)d_ws; size_t off = 0;
  auto alloc = [&](size_t bytes)->void* {
    void* p = wsb + off;
    off += (bytes + 255) & ~(size_t)255;
    return p;
  };

  _Float16* X      = (_Float16*)alloc((size_t)P_*HW_*DPc*2);
  _Float16* cent   = (_Float16*)alloc((size_t)P_*KC*DPc*2);
  float*    cnormb = (float*)   alloc((size_t)P_*KC*4);
  int*      labels = (int*)     alloc((size_t)P_*HW_*4);
  _Float16* encWt  = (_Float16*)alloc((size_t)VIEWS*Fd*DPc*2);
  float*    encB   = (float*)   alloc((size_t)VIEWS*Fd*4);
  float*    feats32= (float*)   alloc((size_t)B_*384*Fd*4);
  _Float16* feats16= (_Float16*)alloc((size_t)B_*384*Fd*2);
  float*    adjraw = (float*)   alloc((size_t)P_*KC*KC*4);
  float*    adjbuf = (float*)   alloc((size_t)P_*KC*KC*4);
  float*    lat0_32= (float*)   alloc((size_t)B_*NLAT*Fd*4);
  _Float16* lat0_16= (_Float16*)alloc((size_t)B_*NLAT*Fd*2);
  float*    lat1_32= (float*)   alloc((size_t)B_*NLAT*Fd*4);
  _Float16* lat1_16= (_Float16*)alloc((size_t)B_*NLAT*Fd*2);
  float*    lat2_32= (float*)   alloc((size_t)B_*NLAT*Fd*4);
  _Float16* lat2_16= (_Float16*)alloc((size_t)B_*NLAT*Fd*2);

  MhaScratch ms;
  ms.wq = (_Float16*)alloc(128*128*2); ms.wk = (_Float16*)alloc(128*128*2);
  ms.wv = (_Float16*)alloc(128*128*2); ms.wo = (_Float16*)alloc(128*128*2);
  ms.q16 = (_Float16*)alloc((size_t)B_*384*128*2);
  ms.k16 = (_Float16*)alloc((size_t)B_*384*128*2);
  ms.v16 = (_Float16*)alloc((size_t)B_*384*128*2);
  ms.vT  = (_Float16*)alloc((size_t)B_*4*32*384*2);
  ms.scores = (float*)alloc((size_t)B_*4*384*32*4);   // max Mq*Mkv = 12288
  ms.att16  = (_Float16*)alloc((size_t)B_*4*384*32*2);
  ms.o16    = (_Float16*)alloc((size_t)B_*384*128*2);
  float* attn_out = (float*)alloc((size_t)B_*384*Fd*4);
  float*    enr32 = (float*)   alloc((size_t)B_*384*Fd*4);
  _Float16* enr16 = (_Float16*)alloc((size_t)B_*384*Fd*2);
  _Float16* W1t   = (_Float16*)alloc(256*128*2);
  _Float16* W2t   = (_Float16*)alloc(128*256*2);
  _Float16* mid16 = (_Float16*)alloc((size_t)B_*384*256*2);
  float*   ffn_out= (float*)   alloc((size_t)B_*384*Fd*4);
  float*   enr2_32= (float*)   alloc((size_t)B_*384*Fd*4);
  _Float16* enr2_16=(_Float16*)alloc((size_t)B_*384*Fd*2);

  GatScratch gs;
  gs.Wt    = (_Float16*)alloc(512*512*2);
  gs.hbuf  = (float*)   alloc((size_t)B_*128*512*4);
  gs.h16   = (_Float16*)alloc((size_t)B_*128*512*2);
  gs.hT    = (_Float16*)alloc((size_t)B_*4*128*128*2);
  gs.a_s   = (float*)   alloc((size_t)B_*128*4*4);
  gs.a_d   = (float*)   alloc((size_t)B_*128*4*4);
  gs.alpha = (_Float16*)alloc((size_t)B_*4*128*128*2);
  float*    x1    = (float*)   alloc((size_t)B_*128*512*4);
  _Float16* x1_16 = (_Float16*)alloc((size_t)B_*128*512*2);
  float*    sc2   = (float*)   alloc((size_t)B_*128*4);
  int*      idx1  = (int*)     alloc((size_t)B_*64*4);
  int*      idx2  = (int*)     alloc((size_t)B_*32*4);
  float*    nx1   = (float*)   alloc((size_t)B_*64*512*4);
  _Float16* nx1_16= (_Float16*)alloc((size_t)B_*64*512*2);
  float*    na1   = (float*)   alloc((size_t)B_*64*64*4);
  float*    x2    = (float*)   alloc((size_t)B_*64*512*4);
  _Float16* x2_16 = (_Float16*)alloc((size_t)B_*64*512*2);
  float*    nx2   = (float*)   alloc((size_t)B_*32*512*4);
  _Float16* nx2_16= (_Float16*)alloc((size_t)B_*32*512*2);
  float*    na2   = (float*)   alloc((size_t)B_*32*32*4);
  float*    x3    = (float*)   alloc((size_t)B_*32*128*4);
  _Float16* x3_16 = (_Float16*)alloc((size_t)B_*32*128*2);
  _Float16* gW1t  = (_Float16*)alloc(64*128*2);
  float*    tbuf  = (float*)   alloc((size_t)B_*32*64*4);
  float*    hview = (float*)   alloc((size_t)3*B_*Fd*4);
  float*    xfu32 = (float*)   alloc((size_t)B_*4*Fd*4);
  _Float16* xfu16 = (_Float16*)alloc((size_t)B_*4*Fd*2);
  _Float16* fwq = (_Float16*)alloc(128*128*2);
  _Float16* fwk = (_Float16*)alloc(128*128*2);
  _Float16* fwv = (_Float16*)alloc(128*128*2);
  _Float16* fwo = (_Float16*)alloc(128*128*2);
  _Float16* qf  = (_Float16*)alloc((size_t)B_*4*128*2);
  _Float16* kf  = (_Float16*)alloc((size_t)B_*4*128*2);
  _Float16* vf  = (_Float16*)alloc((size_t)B_*4*128*2);
  _Float16* of16= (_Float16*)alloc((size_t)B_*4*128*2);
  float*  attn_fu=(float*)   alloc((size_t)B_*4*128*4);
  float*    y32 = (float*)   alloc((size_t)B_*4*128*4);
  _Float16* y16 = (_Float16*)alloc((size_t)B_*4*128*2);
  _Float16* fW1t= (_Float16*)alloc(512*128*2);
  _Float16* fW2t= (_Float16*)alloc(128*512*2);
  _Float16* m16 = (_Float16*)alloc((size_t)B_*4*512*2);
  float*   fout = (float*)   alloc((size_t)B_*4*128*4);
  float*    z32 = (float*)   alloc((size_t)B_*4*128*4);

  // ---- Phase 1: per-view point sets + k-means (WMMA distance GEMMs, L2-resident)
  { long long tot = (long long)P_*HW_*DPc;
    build_comb<<<(int)((tot+255)/256), 256, 0, st>>>(fsag, fcor, faxi, X, tot); }
  cent_init<<<P_*KC, 256, 0, st>>>(X, cent);
  cnorm_kernel<<<P_*KC, 32, 0, st>>>(cent, cnormb);
  for (int it = 0; it < 10; ++it){
    kmeans_assign<<<dim3(HW_/16, P_), 32, 0, st>>>(X, cent, cnormb, labels);
    cent_update<<<dim3(KC, P_), 256, 0, st>>>(X, labels, cent);
    cnorm_kernel<<<P_*KC, 32, 0, st>>>(cent, cnormb);
  }

  // ---- Phase 2: node encoder (cent @ enc_W + b) -> feats (B,384,128), + adjacency
  copy3bias<<<1, 384, 0, st>>>(Pp(1), Pp(3), Pp(5), encB);
  for (int v = 0; v < VIEWS; ++v)
    castT(st, Pp(2*v), 515, Fd, 0, Fd, encWt + (long long)v*Fd*DPc, DPc);
  { GemmP p = gp(); p.nh=3; p.M=KC; p.N=Fd; p.Kd=DPc;
    p.A=cent; p.lda=DPc; p.sAb=(long long)3*KC*DPc; p.sAh=(long long)KC*DPc;
    p.Bt=encWt; p.ldb=DPc; p.sBh=(long long)Fd*DPc;
    p.C=feats32; p.ldc=Fd; p.sCb=(long long)384*Fd; p.sCh_=(long long)KC*Fd;
    p.Ch=feats16; p.ldch=Fd; p.sChb=(long long)384*Fd; p.sChh=(long long)KC*Fd;
    p.bias=encB; p.sBiash=Fd;
    LG(st,p,P_); }
  build_adj_raw<<<P_, KC, 0, st>>>(cent, adjraw);
  { long long tot = (long long)P_*KC*KC;
    adj_sym<<<(int)((tot+255)/256), 256, 0, st>>>(adjraw, adjbuf, tot); }

  // ---- Phase 3: perceiver read/process/write
  { long long tot = (long long)B_*NLAT*Fd;
    bcast_lat<<<(int)((tot+255)/256), 256, 0, st>>>(Pp(6), lat0_32, lat0_16, tot); }
  run_mha(st, NLAT, 384, lat0_16, (long long)NLAT*Fd, feats16, (long long)384*Fd,
          Pp(7), Pp(8), Pp(9), Pp(10), ms, attn_out);
  ln128<<<B_*NLAT, 32, 0, st>>>(attn_out, lat0_32, Pp(11), Pp(12), lat1_32, lat1_16);
  run_mha(st, NLAT, NLAT, lat1_16, (long long)NLAT*Fd, lat1_16, (long long)NLAT*Fd,
          Pp(13), Pp(14), Pp(15), Pp(16), ms, attn_out);
  ln128<<<B_*NLAT, 32, 0, st>>>(attn_out, lat1_32, Pp(17), Pp(18), lat2_32, lat2_16);
  run_mha(st, 384, NLAT, feats16, (long long)384*Fd, lat2_16, (long long)NLAT*Fd,
          Pp(19), Pp(20), Pp(21), Pp(22), ms, attn_out);
  ln128<<<B_*384, 32, 0, st>>>(attn_out, feats32, Pp(23), Pp(24), enr32, enr16);

  // ---- FFN on enriched features
  castT(st, Pp(25), 128, 256, 0, 256, W1t, 128);
  castT(st, Pp(27), 256, 128, 0, 128, W2t, 256);
  { GemmP p = gp(); p.M=384; p.N=256; p.Kd=128;
    p.A=enr16; p.lda=128; p.sAb=(long long)384*128;
    p.Bt=W1t; p.ldb=128;
    p.Ch=mid16; p.ldch=256; p.sChb=(long long)384*256;
    p.bias=Pp(26); p.act=1; LG(st,p,B_); }
  { GemmP p = gp(); p.M=384; p.N=128; p.Kd=256;
    p.A=mid16; p.lda=256; p.sAb=(long long)384*256;
    p.Bt=W2t; p.ldb=256;
    p.C=ffn_out; p.ldc=128; p.sCb=(long long)384*128;
    p.bias=Pp(28); LG(st,p,B_); }
  ln128<<<B_*384, 32, 0, st>>>(ffn_out, enr32, Pp(29), Pp(30), enr2_32, enr2_16);

  // ---- aux head (from processed latents)
  aux_head<<<B_, 32, 0, st>>>(lat2_32, Pp(108), Pp(109), Pp(110), Pp(111), dout + 48);

  // ---- Phase 4: hierarchical GNN per view
  for (int v = 0; v < VIEWS; ++v){
    int gb = 31 + v*20;
    const float* adjv = adjbuf + (long long)v*KC*KC;
    long long adjbs = (long long)VIEWS*KC*KC;
    run_gat(st, 128, 128, 512, 4, enr2_16 + (long long)v*KC*Fd, (long long)384*Fd,
            Pp(gb+0), Pp(gb+1), Pp(gb+2), Pp(gb+3), adjv, adjbs, 1, gs, x1, x1_16);
    run_pool(st, 128, 512, 64, x1, (long long)128*512, adjv, adjbs, Pp(gb+4), Pp(gb+5),
             dout + 96 + (long long)v*B_*KC, idx1, nx1, nx1_16, na1);
    run_gat(st, 64, 512, 512, 4, nx1_16, (long long)64*512,
            Pp(gb+6), Pp(gb+7), Pp(gb+8), Pp(gb+9), na1, (long long)64*64, 1, gs, x2, x2_16);
    run_pool(st, 64, 512, 32, x2, (long long)64*512, na1, (long long)64*64, Pp(gb+10), Pp(gb+11),
             sc2, idx2, nx2, nx2_16, na2);
    run_gat(st, 32, 512, 128, 1, nx2_16, (long long)32*512,
            Pp(gb+12), Pp(gb+13), Pp(gb+14), Pp(gb+15), na2, (long long)32*32, 0, gs, x3, x3_16);
    castT(st, Pp(gb+16), 128, 64, 0, 64, gW1t, 128);
    { GemmP p = gp(); p.M=32; p.N=64; p.Kd=128;
      p.A=x3_16; p.lda=128; p.sAb=(long long)32*128;
      p.Bt=gW1t; p.ldb=128;
      p.C=tbuf; p.ldc=64; p.sCb=(long long)32*64;
      p.bias=Pp(gb+17); p.act=2; LG(st,p,B_); }
    gate_finish<<<B_, 128, 0, st>>>(tbuf, Pp(gb+18), Pp(gb+19), x3, hview + (long long)v*B_*Fd);
  }

  // ---- Phase 5: fusion transformer + classifier
  build_xfu<<<B_, 128, 0, st>>>(Pp(103), hview, hview + (long long)B_*Fd,
                                hview + (long long)2*B_*Fd, xfu32, xfu16);
  castT(st, Pp(91), 128, 384, 0,   128, fwq, 128);
  castT(st, Pp(91), 128, 384, 128, 128, fwk, 128);
  castT(st, Pp(91), 128, 384, 256, 128, fwv, 128);
  castT(st, Pp(93), 128, 128, 0,   128, fwo, 128);
  { GemmP p = gp(); p.M=4; p.N=128; p.Kd=128;
    p.A=xfu16; p.lda=128; p.sAb=4*128;
    p.Bt=fwq; p.ldb=128;
    p.Ch=qf; p.ldch=128; p.sChb=4*128; p.bias=Pp(92); LG(st,p,B_); }
  { GemmP p = gp(); p.M=4; p.N=128; p.Kd=128;
    p.A=xfu16; p.lda=128; p.sAb=4*128;
    p.Bt=fwk; p.ldb=128;
    p.Ch=kf; p.ldch=128; p.sChb=4*128; p.bias=Pp(92)+128; LG(st,p,B_); }
  { GemmP p = gp(); p.M=4; p.N=128; p.Kd=128;
    p.A=xfu16; p.lda=128; p.sAb=4*128;
    p.Bt=fwv; p.ldb=128;
    p.Ch=vf; p.ldch=128; p.sChb=4*128; p.bias=Pp(92)+256; LG(st,p,B_); }
  fusion_attn<<<B_, 32, 0, st>>>(qf, kf, vf, of16);
  { GemmP p = gp(); p.M=4; p.N=128; p.Kd=128;
    p.A=of16; p.lda=128; p.sAb=4*128;
    p.Bt=fwo; p.ldb=128;
    p.C=attn_fu; p.ldc=128; p.sCb=4*128; p.bias=Pp(94); LG(st,p,B_); }
  ln128<<<B_*4, 32, 0, st>>>(attn_fu, xfu32, Pp(95), Pp(96), y32, y16);
  castT(st, Pp(97), 128, 512, 0, 512, fW1t, 128);
  castT(st, Pp(99), 512, 128, 0, 128, fW2t, 512);
  { GemmP p = gp(); p.M=4; p.N=512; p.Kd=128;
    p.A=y16; p.lda=128; p.sAb=4*128;
    p.Bt=fW1t; p.ldb=128;
    p.Ch=m16; p.ldch=512; p.sChb=4*512; p.bias=Pp(98); p.act=1; LG(st,p,B_); }
  { GemmP p = gp(); p.M=4; p.N=128; p.Kd=512;
    p.A=m16; p.lda=512; p.sAb=4*512;
    p.Bt=fW2t; p.ldb=512;
    p.C=fout; p.ldc=128; p.sCb=4*128; p.bias=Pp(100); LG(st,p,B_); }
  ln128<<<B_*4, 32, 0, st>>>(fout, y32, Pp(101), Pp(102), z32, (_Float16*)nullptr);
  cls_head<<<B_, 32, 0, st>>>(z32, Pp(104), Pp(105), Pp(106), Pp(107), dout);
}
